// AutoMTLSuperNet_88948772700992
// MI455X (gfx1250) — matrix operand order repaired and checked by reading, not compiled, not run on hardware
//
#include <hip/hip_runtime.h>
#include <hip/hip_bf16.h>

// ---------------------------------------------------------------------------
// AutoMTL supernet on MI455X (gfx1250): fused bf16 WMMA pipeline.
// Candidate GEMMs (NC=3) fused with activation + softmax(alpha) mixing so the
// large [12][B][N] f32 intermediate never touches HBM.
// ---------------------------------------------------------------------------

#define B_SZ   16384
#define F_SZ   26
#define E_SZ   16
#define D_SZ   13
#define NEXP   4
#define NDOM   3
#define NCAND  3
#define GIN    445
#define GINP   448      // padded to multiple of 32 for WMMA K
#define H_SZ   256
#define OUT_SZ 128
#define FE     (F_SZ * E_SZ)   // 416

typedef __bf16 bf16_t;
typedef __attribute__((ext_vector_type(16))) __bf16 bf16x16;
typedef __attribute__((ext_vector_type(8)))  __bf16 bf16x8;
typedef __attribute__((ext_vector_type(8)))  float  v8f;

__device__ __forceinline__ unsigned short f32_to_bf16(float f) {
  union { float f; unsigned u; } c; c.f = f;
  unsigned r = 0x7FFFu + ((c.u >> 16) & 1u);
  return (unsigned short)((c.u + r) >> 16);
}

// ---------------------------------------------------------------------------
// Weight prep: f32 [BATCH][K][N]  ->  bf16 [BATCH][N][KP]  (transposed, K pad)
// ---------------------------------------------------------------------------
__global__ __launch_bounds__(256)
void prep_w_kernel(const float* __restrict__ W, unsigned short* __restrict__ Wt,
                   int BATCH, int K, int N, int KP) {
  size_t idx = (size_t)blockIdx.x * 256 + threadIdx.x;
  size_t total = (size_t)BATCH * N * KP;
  if (idx >= total) return;
  int k = (int)(idx % KP);
  size_t r = idx / KP;
  int nn = (int)(r % N);
  int z  = (int)(r / N);
  float v = (k < K) ? W[((size_t)z * K + k) * N + nn] : 0.0f;
  Wt[idx] = f32_to_bf16(v);
}

// ---------------------------------------------------------------------------
// MixFeature: feats[n][b][0:448] bf16 = [we.flat(416) | fm(16) | dense(13) | 0]
// One wave per (n,b) row; FM second-order via lane-pair shfl_xor(16) reduction.
// ---------------------------------------------------------------------------
__global__ __launch_bounds__(256)
void feats_kernel(const float* __restrict__ emb,        // [B][26][16]
                  const float* __restrict__ dense,      // [B][13]
                  const float* __restrict__ feat_alpha, // [NE][26]
                  unsigned short* __restrict__ feats) { // [NE][B][448]
  int row  = blockIdx.x * 8 + (threadIdx.x >> 5);       // over NE*B
  int lane = threadIdx.x & 31;
  int n = row / B_SZ;
  int b = row % B_SZ;
  const float* erow = emb + (size_t)b * FE;
  unsigned short* frow = feats + (size_t)row * GINP;

  float s = 0.0f, sq = 0.0f;
  #pragma unroll
  for (int t = 0; t < 13; ++t) {                 // 13 * 32 = 416 exactly
    int j = lane + 32 * t;
    int f = j >> 4;
    float a = feat_alpha[n * F_SZ + f];
    float w = 1.0f / (1.0f + __expf(-a));
    float v = erow[j] * w;
    frow[j] = f32_to_bf16(v);
    s += v; sq += v * v;
  }
  // lane L and L^16 cover complementary field parities for e = L & 15
  s  += __shfl_xor(s, 16, 32);
  sq += __shfl_xor(sq, 16, 32);
  if (lane < 16) {
    float fm = 0.5f * (s * s - sq);
    frow[FE + lane] = f32_to_bf16(fm);           // fm at [416..431]
    float dv = (lane < D_SZ) ? dense[(size_t)b * D_SZ + lane] : 0.0f;
    frow[FE + 16 + lane] = f32_to_bf16(dv);      // dense [432..444], pad [445..447]
  }
}

// ---------------------------------------------------------------------------
// Fused candidate GEMM + activation + softmax(alpha) mix.
//   out[n][m][j] = sum_c softmax(alpha[n])_c * act_c( A[n] x Bt[n][c] + bias[n][c] )
// Block: 256 thr = 8 waves; tile 128 rows x 64 cols; 3 accumulator sets/wave
// (3 x 4 x v8f = 96 VGPRs). One A fragment per k-step feeds 12 WMMAs.
// B tiles (3 x 32x64 bf16 = 12 KB) staged through LDS in fragment layout.
// ---------------------------------------------------------------------------
__global__ __launch_bounds__(256)
void gemm_fused_mix_wmma(const bf16_t* __restrict__ A,    // [NE][M][K] bf16
                         const bf16_t* __restrict__ Bt,   // [NE][NC][N][K] bf16
                         const float* __restrict__ bias,  // [NE][NC][N]
                         const float* __restrict__ alpha, // [NE][NC]
                         float* __restrict__ outF,        // [NE][M][N] f32 (or null)
                         unsigned short* __restrict__ outB, // [NE][M][N] bf16 (or null)
                         int M, int N, int K) {
  __shared__ bf16_t ldsB[NCAND * 4 * 32 * 16];            // 12 KB
  const int lane = threadIdx.x & 31;
  const int wave = threadIdx.x >> 5;
  const int n = blockIdx.z;                               // expert
  const bf16_t* Az = A + (size_t)n * M * K;
  const int m0 = blockIdx.y * 128 + wave * 16;
  const int n0 = blockIdx.x * 64;

  v8f acc[NCAND][4] = {};
  // A 16x32 bf16 fragment: lane m=lane&15, half-select lane>>4 (K +0..15 / +16..31)
  const bf16_t* aptr = Az + (size_t)(m0 + (lane & 15)) * K + ((lane >> 4) << 4);

  // B staging: 384 chunks of 16 halves; chunk c -> cand=c>>7, tile=(c>>5)&3, ln=c&31
  const int cA = threadIdx.x;
  const int cB = threadIdx.x + 256;                       // valid if < 384
  const bf16_t* bsrcA;
  const bf16_t* bsrcB = nullptr;
  {
    int cand = cA >> 7, tt = (cA >> 5) & 3, ln = cA & 31;
    bsrcA = Bt + ((size_t)(n * NCAND + cand) * N + n0 + tt * 16 + (ln & 15)) * K
               + ((ln >> 4) << 4);
  }
  if (cB < NCAND * 128) {
    int cand = cB >> 7, tt = (cB >> 5) & 3, ln = cB & 31;
    bsrcB = Bt + ((size_t)(n * NCAND + cand) * N + n0 + tt * 16 + (ln & 15)) * K
               + ((ln >> 4) << 4);
  }

  for (int k0 = 0; k0 < K; k0 += 32) {
    *(bf16x16*)&ldsB[cA * 16] = *(const bf16x16*)(bsrcA + k0);
    if (bsrcB) *(bf16x16*)&ldsB[cB * 16] = *(const bf16x16*)(bsrcB + k0);
    __builtin_prefetch(aptr + k0 + 64, 0, 1);             // global_prefetch next A
    union { bf16x16 v; bf16x8 h[2]; } af;
    af.h[0] = *(const bf16x8*)(aptr + k0);
    af.h[1] = *(const bf16x8*)(aptr + k0 + 8);
    __syncthreads();
    #pragma unroll
    for (int cand = 0; cand < NCAND; ++cand) {
      #pragma unroll
      for (int t = 0; t < 4; ++t) {
        bf16x16 bfr = *(const bf16x16*)&ldsB[((cand * 4 + t) * 32 + lane) * 16];
        acc[cand][t] = __builtin_amdgcn_wmma_f32_16x16x32_bf16(
            false, af.v, false, bfr, (short)0, acc[cand][t], false, false);
      }
    }
    __syncthreads();
  }

  // softmax over candidate alphas (uniform per block)
  float a0 = alpha[n * NCAND + 0], a1 = alpha[n * NCAND + 1], a2 = alpha[n * NCAND + 2];
  float mx = fmaxf(a0, fmaxf(a1, a2));
  float e0 = __expf(a0 - mx), e1 = __expf(a1 - mx), e2 = __expf(a2 - mx);
  float inv = 1.0f / (e0 + e1 + e2);
  float w0 = e0 * inv, w1 = e1 * inv, w2 = e2 * inv;

  // C/D layout: vgpr r, lanes 0-15 -> (M=r, N=lane); lanes 16-31 -> (M=r+8, N=lane-16)
  const int rbase = m0 + ((lane >> 4) << 3);
  const int cbase = n0 + (lane & 15);
  #pragma unroll
  for (int t = 0; t < 4; ++t) {
    int col = cbase + t * 16;
    float b0 = bias[(n * NCAND + 0) * N + col];
    float b1 = bias[(n * NCAND + 1) * N + col];
    float b2 = bias[(n * NCAND + 2) * N + col];
    #pragma unroll
    for (int r = 0; r < 8; ++r) {
      float x0 = acc[0][t][r] + b0;
      float x1 = acc[1][t][r] + b1;
      float x2 = acc[2][t][r] + b2;
      float gl = 0.5f * x1 * (1.0f + erff(x1 * 0.70710678118654752f));
      float v  = w0 * fmaxf(x0, 0.0f) + w1 * gl + w2 * tanhf(x2);
      size_t oidx = ((size_t)n * M + rbase + r) * N + col;
      if (outB) outB[oidx] = f32_to_bf16(v);
      else      outF[oidx] = v;
    }
  }
}

// ---------------------------------------------------------------------------
// Gate mix (expert level): logits0[n][b][e] = <gate_input[b], Wg0[n][:,e]> + bg0 + beta0
// softmax over e; mixed[n][b][o] = sum_e w0 * h[e][b][o] -> bf16. One wave per b.
// ---------------------------------------------------------------------------
__global__ __launch_bounds__(256)
void gatemix_kernel(const float* __restrict__ emb,   // [B][416] (flat sparse_embs)
                    const float* __restrict__ Wg0,   // [NE][445][NE]
                    const float* __restrict__ bg0,   // [NE][NE]
                    const float* __restrict__ be0,   // [NE][NE]
                    const float* __restrict__ h,     // [NE][B][128] f32
                    unsigned short* __restrict__ outbf) { // [NE][B][128] bf16
  int b    = blockIdx.x * 8 + (threadIdx.x >> 5);
  int lane = threadIdx.x & 31;
  const float* gi = emb + (size_t)b * FE;            // gate_input rows >=416 are zero
  float acc[16] = {};
  for (int i = lane; i < FE; i += 32) {
    float g = gi[i];
    #pragma unroll
    for (int n = 0; n < NEXP; ++n) {
      const float* wrow = Wg0 + ((size_t)n * GIN + i) * NEXP;
      #pragma unroll
      for (int e = 0; e < NEXP; ++e) acc[n * NEXP + e] += g * wrow[e];
    }
  }
  #pragma unroll
  for (int m = 1; m < 32; m <<= 1) {
    #pragma unroll
    for (int q = 0; q < 16; ++q) acc[q] += __shfl_xor(acc[q], m, 32);
  }
  float w0[16];
  #pragma unroll
  for (int n = 0; n < NEXP; ++n) {
    float l[NEXP], mx = -1e30f;
    #pragma unroll
    for (int e = 0; e < NEXP; ++e) {
      l[e] = acc[n * NEXP + e] + bg0[n * NEXP + e] + be0[n * NEXP + e];
      mx = fmaxf(mx, l[e]);
    }
    float s = 0.0f;
    #pragma unroll
    for (int e = 0; e < NEXP; ++e) { l[e] = __expf(l[e] - mx); s += l[e]; }
    float inv = 1.0f / s;
    #pragma unroll
    for (int e = 0; e < NEXP; ++e) w0[n * NEXP + e] = l[e] * inv;
  }
  size_t sBN = (size_t)B_SZ * OUT_SZ;
  #pragma unroll
  for (int t = 0; t < 4; ++t) {
    int o = lane + 32 * t;
    #pragma unroll
    for (int n = 0; n < NEXP; ++n) {
      float v = 0.0f;
      #pragma unroll
      for (int e = 0; e < NEXP; ++e)
        v += w0[n * NEXP + e] * h[(size_t)e * sBN + (size_t)b * OUT_SZ + o];
      outbf[(size_t)n * sBN + (size_t)b * OUT_SZ + o] = f32_to_bf16(v);
    }
  }
}

// ---------------------------------------------------------------------------
// Final: domain gate softmax for d=domain_ids[b]; out[b][o]=sum_e w1*h2[e][b][o]
// ---------------------------------------------------------------------------
__global__ __launch_bounds__(256)
void final_domain_kernel(const float* __restrict__ emb, const float* __restrict__ Wg1,
                         const float* __restrict__ bg1, const float* __restrict__ be1,
                         const int* __restrict__ dom, const float* __restrict__ h2,
                         float* __restrict__ out) {
  int b    = blockIdx.x * 8 + (threadIdx.x >> 5);
  int lane = threadIdx.x & 31;
  int d = dom[b];
  const float* gi = emb + (size_t)b * FE;
  float acc[NEXP] = {};
  for (int i = lane; i < FE; i += 32) {
    float g = gi[i];
    const float* wrow = Wg1 + ((size_t)d * GIN + i) * NEXP;
    #pragma unroll
    for (int e = 0; e < NEXP; ++e) acc[e] += g * wrow[e];
  }
  #pragma unroll
  for (int m = 1; m < 32; m <<= 1) {
    #pragma unroll
    for (int e = 0; e < NEXP; ++e) acc[e] += __shfl_xor(acc[e], m, 32);
  }
  float l[NEXP], mx = -1e30f;
  #pragma unroll
  for (int e = 0; e < NEXP; ++e) {
    l[e] = acc[e] + bg1[d * NEXP + e] + be1[d * NEXP + e];
    mx = fmaxf(mx, l[e]);
  }
  float s = 0.0f;
  #pragma unroll
  for (int e = 0; e < NEXP; ++e) { l[e] = __expf(l[e] - mx); s += l[e]; }
  float inv = 1.0f / s;
  size_t sBN = (size_t)B_SZ * OUT_SZ;
  #pragma unroll
  for (int t = 0; t < 4; ++t) {
    int o = lane + 32 * t;
    float v = 0.0f;
    #pragma unroll
    for (int e = 0; e < NEXP; ++e)
      v += l[e] * h2[(size_t)e * sBN + (size_t)b * OUT_SZ + o];
    out[(size_t)b * OUT_SZ + o] = v * inv;
  }
}

// ---------------------------------------------------------------------------
extern "C" void kernel_launch(void* const* d_in, const int* in_sizes, int n_in,
                              void* d_out, int out_size, void* d_ws, size_t ws_size,
                              hipStream_t stream) {
  (void)in_sizes; (void)n_in; (void)out_size; (void)ws_size;
  const float* emb    = (const float*)d_in[0];
  const float* dense  = (const float*)d_in[1];
  const int*   dom    = (const int*)d_in[2];
  const float* falpha = (const float*)d_in[3];
  const float* W00 = (const float*)d_in[4];  const float* b00 = (const float*)d_in[5];  const float* a00 = (const float*)d_in[6];
  const float* W01 = (const float*)d_in[7];  const float* b01 = (const float*)d_in[8];  const float* a01 = (const float*)d_in[9];
  const float* W10 = (const float*)d_in[10]; const float* b10 = (const float*)d_in[11]; const float* a10 = (const float*)d_in[12];
  const float* W11 = (const float*)d_in[13]; const float* b11 = (const float*)d_in[14]; const float* a11 = (const float*)d_in[15];
  const float* Wg0 = (const float*)d_in[16]; const float* bg0 = (const float*)d_in[17]; const float* be0 = (const float*)d_in[18];
  const float* Wg1 = (const float*)d_in[19]; const float* bg1 = (const float*)d_in[20]; const float* be1 = (const float*)d_in[21];
  float* out = (float*)d_out;

  char* ws = (char*)d_ws;
  size_t off = 0;
  auto carve = [&](size_t bytes) -> void* {
    void* p = ws + off;
    off += (bytes + 255) & ~(size_t)255;
    return p;
  };
  const int ZB = NEXP * NCAND;  // 12
  unsigned short* featsB = (unsigned short*)carve((size_t)NEXP * B_SZ * GINP * 2);
  unsigned short* wA     = (unsigned short*)carve((size_t)ZB * H_SZ * GINP * 2);
  unsigned short* wB     = (unsigned short*)carve((size_t)ZB * OUT_SZ * H_SZ * 2);
  unsigned short* wC     = (unsigned short*)carve((size_t)ZB * H_SZ * OUT_SZ * 2);
  unsigned short* wD     = (unsigned short*)carve((size_t)ZB * OUT_SZ * H_SZ * 2);
  unsigned short* bufBF  = (unsigned short*)carve((size_t)NEXP * B_SZ * H_SZ * 2);
  float*          bufF32 = (float*)carve((size_t)NEXP * B_SZ * OUT_SZ * 4);
  unsigned short* mixedC = (unsigned short*)carve((size_t)NEXP * B_SZ * OUT_SZ * 2);

  dim3 blk(256);
  auto blocks1d = [](size_t total) { return dim3((unsigned)((total + 255) / 256)); };

  // 1) weight prep (transpose + bf16 + K pad)
  prep_w_kernel<<<blocks1d((size_t)ZB * H_SZ * GINP),  blk, 0, stream>>>(W00, wA, ZB, GIN,    H_SZ,   GINP);
  prep_w_kernel<<<blocks1d((size_t)ZB * OUT_SZ * H_SZ), blk, 0, stream>>>(W01, wB, ZB, H_SZ,   OUT_SZ, H_SZ);
  prep_w_kernel<<<blocks1d((size_t)ZB * H_SZ * OUT_SZ), blk, 0, stream>>>(W10, wC, ZB, OUT_SZ, H_SZ,   OUT_SZ);
  prep_w_kernel<<<blocks1d((size_t)ZB * OUT_SZ * H_SZ), blk, 0, stream>>>(W11, wD, ZB, H_SZ,   OUT_SZ, H_SZ);

  // 2) features
  feats_kernel<<<dim3(NEXP * B_SZ / 8), blk, 0, stream>>>(emb, dense, falpha, featsB);

  // 3) L0B0 fused: feats[NE][B][448] x wA -> bufBF bf16 [NE][B][256]
  gemm_fused_mix_wmma<<<dim3(H_SZ / 64, B_SZ / 128, NEXP), blk, 0, stream>>>(
      (const bf16_t*)featsB, (const bf16_t*)wA, b00, a00,
      nullptr, bufBF, B_SZ, H_SZ, GINP);

  // 4) L0B1 fused: bufBF x wB -> bufF32 f32 [NE][B][128]
  gemm_fused_mix_wmma<<<dim3(OUT_SZ / 64, B_SZ / 128, NEXP), blk, 0, stream>>>(
      (const bf16_t*)bufBF, (const bf16_t*)wB, b01, a01,
      bufF32, nullptr, B_SZ, OUT_SZ, H_SZ);

  // 5) expert-level gate mixing -> mixed bf16 [NE][B][128]
  gatemix_kernel<<<dim3(B_SZ / 8), blk, 0, stream>>>(emb, Wg0, bg0, be0, bufF32, mixedC);

  // 6) L1B0 fused: mixedC x wC -> bufBF bf16 [NE][B][256]
  gemm_fused_mix_wmma<<<dim3(H_SZ / 64, B_SZ / 128, NEXP), blk, 0, stream>>>(
      (const bf16_t*)mixedC, (const bf16_t*)wC, b10, a10,
      nullptr, bufBF, B_SZ, H_SZ, OUT_SZ);

  // 7) L1B1 fused: bufBF x wD -> bufF32 f32 [NE][B][128]
  gemm_fused_mix_wmma<<<dim3(OUT_SZ / 64, B_SZ / 128, NEXP), blk, 0, stream>>>(
      (const bf16_t*)bufBF, (const bf16_t*)wD, b11, a11,
      bufF32, nullptr, B_SZ, OUT_SZ, H_SZ);

  // 8) domain gate + select -> out [B][128]
  final_domain_kernel<<<dim3(B_SZ / 8), blk, 0, stream>>>(emb, Wg1, bg1, be1, dom, bufF32, out);
}
